// BertMTL1_23897198035038
// MI455X (gfx1250) — matrix operand order repaired and checked by reading, not compiled, not run on hardware
//
#include <hip/hip_runtime.h>

// ---------------- problem constants ----------------
#define BB 4
#define SS 512
#define NN 128
#define EE 128
#define HH 120
#define RR 97
#define DD 768
#define LL 12
#define NHEAD 12
#define DHEAD 64
#define FFD 3072

typedef __attribute__((ext_vector_type(16))) _Float16 v16h;
typedef __attribute__((ext_vector_type(8)))  float    v8f;
typedef __attribute__((ext_vector_type(4)))  float    f4;
typedef __attribute__((ext_vector_type(4)))  _Float16 h4;

// ---------------- WMMA GEMM tile config ----------------
#define BM 128
#define BN 64
#define BK 32
#define ALD 40   // As row stride in halves (rows 16B aligned: 40*2=80)
#define BLD 72   // Bs row stride in halves (rows 16B aligned: 72*2=144)

struct GemmP {
  const void* A;       // row-major M x K (lda)
  const void* Bm;      // bT=0: K x N (ldb); bT=1: N x K (ldb)
  float*      C;       // optional fp32 out
  _Float16*   Ch;      // optional f16 out (same indexing)
  const float* bias;   // optional per-N bias
  long long divA, bsA1, bsA2;      // offA = (z/divA)*bsA1 + (z%divA)*bsA2
  long long divB, bsB1, bsB2;
  long long divC, bsC1, bsC2;
  long long divBias, bsBias1, bsBias2;
  int lda, ldb, ldc, cstride, bias_estride;
  int M, N, K;
  float alpha;
  int act;             // 0 none, 1 exact gelu, 2 tanh
};

template <int AF16, int BF16, int BT>
__global__ __launch_bounds__(256) void gemm_wmma_kernel(GemmP p) {
  __shared__ _Float16 As[BM * ALD];
  __shared__ _Float16 Bs[BK * BLD];
  const int z = blockIdx.z;
  const long long offA = (z / p.divA) * p.bsA1 + (z % p.divA) * p.bsA2;
  const long long offB = (z / p.divB) * p.bsB1 + (z % p.divB) * p.bsB2;
  const float*    Af = (const float*)p.A + offA;
  const _Float16* Ah = (const _Float16*)p.A + offA;
  const float*    Bf = (const float*)p.Bm + offB;
  const _Float16* Bh = (const _Float16*)p.Bm + offB;
  const int m0 = blockIdx.y * BM;
  const int n0 = blockIdx.x * BN;
  const int tid = threadIdx.x;
  const int wave = tid >> 5, lane = tid & 31;
  const int wm = wave & 3, wn = wave >> 2;     // 4x2 wave grid -> 128x64
  const int hs = lane >> 4, l16 = lane & 15;
  const bool aFull = (m0 + BM <= p.M);
  const bool bFull = (n0 + BN <= p.N);
  v8f acc[2][2];
  for (int i = 0; i < 2; ++i)
    for (int j = 0; j < 2; ++j)
      for (int u = 0; u < 8; ++u) acc[i][j][u] = 0.f;

  const int nK = (p.K + BK - 1) / BK;
  for (int kc = 0; kc < nK; ++kc) {
    const int k0 = kc * BK;
    const bool kFull = (k0 + BK <= p.K);
    // speculative prefetch of next K-chunk (global_prefetch_b8)
    if (kc + 1 < nK && aFull && bFull) {
      const int kk = (tid & 7) * 4;
      const int mb = tid >> 3;
      if (AF16) __builtin_prefetch(Ah + (long long)(m0 + mb) * p.lda + k0 + BK + kk, 0, 1);
      else      __builtin_prefetch(Af + (long long)(m0 + mb) * p.lda + k0 + BK + kk, 0, 1);
    }
    // ---- stage A tile (BM x BK) ----
    if (aFull && kFull) {
      const int kk = (tid & 7) * 4;
      const int mb = tid >> 3;
      #pragma unroll
      for (int pass = 0; pass < 4; ++pass) {
        const int m = mb + pass * 32;
        h4 hv;
        if (AF16) {
          hv = *(const h4*)(Ah + (long long)(m0 + m) * p.lda + k0 + kk);
        } else {
          f4 fv = *(const f4*)(Af + (long long)(m0 + m) * p.lda + k0 + kk);
          hv[0] = (_Float16)fv[0]; hv[1] = (_Float16)fv[1];
          hv[2] = (_Float16)fv[2]; hv[3] = (_Float16)fv[3];
        }
        *(h4*)(&As[m * ALD + kk]) = hv;
      }
    } else {
      // clamp-address + select-zero: branchless edge handling
      const int kk = (tid & 7) * 4;
      const int mb = tid >> 3;
      #pragma unroll
      for (int pass = 0; pass < 4; ++pass) {
        const int m = mb + pass * 32;
        const int gm = m0 + m;
        const int gmc = gm < p.M ? gm : p.M - 1;
        #pragma unroll
        for (int u = 0; u < 4; ++u) {
          const int gk = k0 + kk + u;
          const int gkc = gk < p.K ? gk : p.K - 1;
          float v = AF16 ? (float)Ah[(long long)gmc * p.lda + gkc]
                         : Af[(long long)gmc * p.lda + gkc];
          v = (gm < p.M && gk < p.K) ? v : 0.f;
          As[m * ALD + kk + u] = (_Float16)v;
        }
      }
    }
    // ---- stage B tile (BK x BN) ----
    if (BT == 0) {
      if (bFull && kFull) {
        const int nn = (tid & 15) * 4;
        const int kb = tid >> 4;
        #pragma unroll
        for (int pass = 0; pass < 2; ++pass) {
          const int k = kb + pass * 16;
          h4 hv;
          if (BF16) {
            hv = *(const h4*)(Bh + (long long)(k0 + k) * p.ldb + n0 + nn);
          } else {
            f4 fv = *(const f4*)(Bf + (long long)(k0 + k) * p.ldb + n0 + nn);
            hv[0] = (_Float16)fv[0]; hv[1] = (_Float16)fv[1];
            hv[2] = (_Float16)fv[2]; hv[3] = (_Float16)fv[3];
          }
          *(h4*)(&Bs[k * BLD + nn]) = hv;
        }
      } else {
        const int nn = (tid & 15) * 4;
        const int kb = tid >> 4;
        #pragma unroll
        for (int pass = 0; pass < 2; ++pass) {
          const int k = kb + pass * 16;
          const int gk = k0 + k;
          const int gkc = gk < p.K ? gk : p.K - 1;
          #pragma unroll
          for (int u = 0; u < 4; ++u) {
            const int gn = n0 + nn + u;
            const int gnc = gn < p.N ? gn : p.N - 1;
            float v = BF16 ? (float)Bh[(long long)gkc * p.ldb + gnc]
                           : Bf[(long long)gkc * p.ldb + gnc];
            v = (gk < p.K && gn < p.N) ? v : 0.f;
            Bs[k * BLD + nn + u] = (_Float16)v;
          }
        }
      }
    } else {  // BT == 1: B is N x K; vector-load along K, scatter-transpose into LDS
      if (bFull && kFull) {
        const int n = tid & 63;
        const int kk = (tid >> 6) * 4;
        #pragma unroll
        for (int pass = 0; pass < 2; ++pass) {
          const int kb = kk + pass * 16;
          h4 hv;
          if (BF16) {
            hv = *(const h4*)(Bh + (long long)(n0 + n) * p.ldb + k0 + kb);
          } else {
            f4 fv = *(const f4*)(Bf + (long long)(n0 + n) * p.ldb + k0 + kb);
            hv[0] = (_Float16)fv[0]; hv[1] = (_Float16)fv[1];
            hv[2] = (_Float16)fv[2]; hv[3] = (_Float16)fv[3];
          }
          #pragma unroll
          for (int u = 0; u < 4; ++u) Bs[(kb + u) * BLD + n] = hv[u];
        }
      } else {
        const int n = tid & 63;
        const int kk = (tid >> 6) * 4;
        const int gn = n0 + n;
        const int gnc = gn < p.N ? gn : p.N - 1;
        #pragma unroll
        for (int pass = 0; pass < 2; ++pass) {
          const int kb = kk + pass * 16;
          #pragma unroll
          for (int u = 0; u < 4; ++u) {
            const int gk = k0 + kb + u;
            const int gkc = gk < p.K ? gk : p.K - 1;
            float v = BF16 ? (float)Bh[(long long)gnc * p.ldb + gkc]
                           : Bf[(long long)gnc * p.ldb + gkc];
            v = (gk < p.K && gn < p.N) ? v : 0.f;
            Bs[(kb + u) * BLD + n] = (_Float16)v;
          }
        }
      }
    }
    __syncthreads();
    // ---- fragments per ISA 7.12.2 layouts ----
    v16h afr[2], bfr[2];
    #pragma unroll
    for (int i = 0; i < 2; ++i) {
      const _Float16* ap = &As[(wm * 32 + i * 16 + l16) * ALD + hs * 8];
      #pragma unroll
      for (int u = 0; u < 8; ++u) { afr[i][u] = ap[u]; afr[i][u + 8] = ap[16 + u]; }
    }
    #pragma unroll
    for (int j = 0; j < 2; ++j) {
      const _Float16* bp = &Bs[(hs * 8) * BLD + wn * 32 + j * 16 + l16];
      #pragma unroll
      for (int u = 0; u < 8; ++u) { bfr[j][u] = bp[u * BLD]; bfr[j][u + 8] = bp[(16 + u) * BLD]; }
    }
    #pragma unroll
    for (int i = 0; i < 2; ++i)
      #pragma unroll
      for (int j = 0; j < 2; ++j)
        acc[i][j] = __builtin_amdgcn_wmma_f32_16x16x32_f16(
            false, afr[i], false, bfr[j], (short)0, acc[i][j], false, false);
    __syncthreads();
  }
  // ---- epilogue: alpha, bias, activation, fp32/f16 stores ----
  const long long offC    = (z / p.divC) * p.bsC1 + (z % p.divC) * p.bsC2;
  const long long offBias = (z / p.divBias) * p.bsBias1 + (z % p.divBias) * p.bsBias2;
  for (int i = 0; i < 2; ++i)
    for (int j = 0; j < 2; ++j) {
      const int n = n0 + wn * 32 + j * 16 + l16;
      if (n >= p.N) continue;
      float bvv = 0.f;
      if (p.bias) bvv = p.bias[offBias + (long long)n * p.bias_estride];
      for (int r = 0; r < 8; ++r) {
        const int m = m0 + wm * 32 + i * 16 + hs * 8 + r;
        if (m >= p.M) continue;
        float v = acc[i][j][r] * p.alpha + bvv;
        if (p.act == 1) v = 0.5f * v * (1.0f + erff(v * 0.70710678118f));
        else if (p.act == 2) v = tanhf(v);
        const long long ci = offC + (long long)m * p.ldc + (long long)n * p.cstride;
        if (p.C)  p.C[ci]  = v;
        if (p.Ch) p.Ch[ci] = (_Float16)v;
      }
    }
}

// ---------------- helpers / elementwise kernels ----------------
__device__ inline float block_reduce_sum(float v, float* red) {
  int tid = threadIdx.x;
  red[tid] = v; __syncthreads();
  for (int s = blockDim.x >> 1; s > 0; s >>= 1) {
    if (tid < s) red[tid] += red[tid + s];
    __syncthreads();
  }
  float r = red[0]; __syncthreads();
  return r;
}

__global__ void embed_kernel(const int* ids, const float* masks,
                             const float* tok, const float* pos, const float* typ,
                             const float* g, const float* bv,
                             float* x, _Float16* xh, float* sbias) {
  __shared__ float red[256];
  const int row = blockIdx.x;
  const int s = row % SS;
  const int id = ids[row];
  const int tid = threadIdx.x;
  float v[3]; float loc = 0.f;
  for (int t = 0; t < 3; ++t) {
    int d = tid + t * 256;
    v[t] = tok[(size_t)id * DD + d] + pos[(size_t)s * DD + d] + typ[d];
    loc += v[t];
  }
  float mean = block_reduce_sum(loc, red) * (1.f / DD);
  loc = 0.f;
  for (int t = 0; t < 3; ++t) { float dd = v[t] - mean; loc += dd * dd; }
  float var = block_reduce_sum(loc, red) * (1.f / DD);
  float rstd = rsqrtf(var + 1e-12f);
  for (int t = 0; t < 3; ++t) {
    int d = tid + t * 256;
    float o = (v[t] - mean) * rstd * g[d] + bv[d];
    x[(size_t)row * DD + d] = o;
    xh[(size_t)row * DD + d] = (_Float16)o;
  }
  if (tid == 0) sbias[row] = (1.f - masks[row]) * -10000.f;
}

__global__ void addln_kernel(const float* xin, const float* dlt,
                             const float* g, const float* bv,
                             float* xo, _Float16* xho) {
  __shared__ float red[256];
  const int row = blockIdx.x, tid = threadIdx.x;
  float v[3]; float loc = 0.f;
  for (int t = 0; t < 3; ++t) {
    size_t i = (size_t)row * DD + tid + t * 256;
    v[t] = xin[i] + dlt[i];
    loc += v[t];
  }
  float mean = block_reduce_sum(loc, red) * (1.f / DD);
  loc = 0.f;
  for (int t = 0; t < 3; ++t) { float d = v[t] - mean; loc += d * d; }
  float var = block_reduce_sum(loc, red) * (1.f / DD);
  float rstd = rsqrtf(var + 1e-12f);
  for (int t = 0; t < 3; ++t) {
    int d = tid + t * 256;
    size_t i = (size_t)row * DD + d;
    float o = (v[t] - mean) * rstd * g[d] + bv[d];
    xo[i] = o;
    xho[i] = (_Float16)o;
  }
}

__global__ void softmax_kernel(float* sc) {
  __shared__ float red[256];
  const int tid = threadIdx.x;
  float* p = sc + (size_t)blockIdx.x * SS;
  float v0 = p[tid], v1 = p[tid + 256];
  red[tid] = fmaxf(v0, v1); __syncthreads();
  for (int s = 128; s > 0; s >>= 1) {
    if (tid < s) red[tid] = fmaxf(red[tid], red[tid + s]);
    __syncthreads();
  }
  float mx = red[0]; __syncthreads();
  float e0 = __expf(v0 - mx), e1 = __expf(v1 - mx);
  float sum = block_reduce_sum(e0 + e1, red);
  float inv = 1.f / sum;
  p[tid] = e0 * inv; p[tid + 256] = e1 * inv;
}

__global__ void counts_kernel(const int* m, int* counts, int* cmax, int nb, int len) {
  const int tid = threadIdx.x;
  if (tid < nb) {
    int c = 0;
    for (int i = 0; i < len; ++i) c += (m[tid * len + i] == 1);
    counts[tid] = c;
  }
  __syncthreads();
  if (tid == 0) {
    int mm = 0;
    for (int b = 0; b < nb; ++b) mm = counts[b] > mm ? counts[b] : mm;
    *cmax = mm;
  }
}

// stable partition order (argsort of 1-mask, stable) via prefix scan; blockDim==len
__global__ void scan_scatter_kernel(const int* m, int* order, int len) {
  __shared__ int sc[512];
  const int b = blockIdx.x, i = threadIdx.x;
  const int v = (m[b * len + i] == 1);
  sc[i] = v; __syncthreads();
  for (int off = 1; off < len; off <<= 1) {
    int t = (i >= off) ? sc[i - off] : 0;
    __syncthreads();
    sc[i] += t;
    __syncthreads();
  }
  const int ones = sc[len - 1];
  const int pre1 = sc[i] - v;
  const int dest = v ? pre1 : ones + (i - pre1);
  order[b * len + dest] = i;
}

__global__ void gather_co_kernel(const float* x, const int* counts, const int* cmax,
                                 const int* order, float* co2) {
  const int blk = blockIdx.x;
  const int b = blk / SS, j = blk % SS;
  const int tid = threadIdx.x;
  float* o = co2 + (size_t)blk * DD;
  if (j < counts[b]) {
    const int src = order[b * SS + j];
    const float* r = x + ((size_t)b * SS + src) * DD;
    for (int t = 0; t < 3; ++t) { int d = tid + t * 256; o[d] = r[d]; }
  } else {
    const float pv = (j < *cmax) ? -1.f : 0.f;
    for (int t = 0; t < 3; ++t) { int d = tid + t * 256; o[d] = pv; }
  }
}

__global__ void pexp_kernel(const float* bil, float* P) {
  const int idx = blockIdx.x * blockDim.x + threadIdx.x;
  if (idx >= BB * NN * NN) return;
  const int r = idx % (NN * NN);
  const int i = r / NN, j = r % NN;
  P[idx] = (i == j) ? 0.f : __expf(bil[idx]);
}

__global__ void colsum_kernel(const float* P, float* cs) {
  __shared__ float red[128];
  const int b = blockIdx.x / NN, j = blockIdx.x % NN, tid = threadIdx.x;
  red[tid] = P[((size_t)b * NN + tid) * NN + j];
  __syncthreads();
  for (int s = 64; s > 0; s >>= 1) {
    if (tid < s) red[tid] += red[tid + s];
    __syncthreads();
  }
  if (tid == 0) cs[blockIdx.x] = red[0];
}

__global__ void aug_build_kernel(const float* P, const float* cs, const float* root, float* aug) {
  const int b = blockIdx.x / NN, i = blockIdx.x % NN, j = threadIdx.x;
  float v;
  if (j < NN) {
    if (i == 1) v = root[b * NN + j];
    else if (i == j) v = cs[b * NN + j];
    else v = -P[((size_t)b * NN + i) * NN + j];
  } else {
    v = ((j - NN) == i) ? 1.f : 0.f;
  }
  aug[((size_t)b * NN + i) * 256 + j] = v;
}

__global__ void gj_kernel(float* aug) {  // one block (256 thr) per batch
  float* A = aug + (size_t)blockIdx.x * NN * 256;
  __shared__ float colk[NN];
  const int tid = threadIdx.x;
  for (int k = 0; k < NN; ++k) {
    if (tid < NN) colk[tid] = A[(size_t)tid * 256 + k];
    __syncthreads();
    const float ip = 1.f / colk[k];
    A[(size_t)k * 256 + tid] *= ip;
    __syncthreads();
    const float pr = A[(size_t)k * 256 + tid];
    for (int i = 0; i < NN; ++i)
      if (i != k) A[(size_t)i * 256 + tid] -= colk[i] * pr;
    __syncthreads();
  }
}

__global__ void invcopy_kernel(const float* aug, float* inv) {
  const int idx = blockIdx.x * blockDim.x + threadIdx.x;
  if (idx >= BB * NN * NN) return;
  const int b = idx / (NN * NN), r = idx % (NN * NN);
  const int i = r / NN, j = r % NN;
  inv[idx] = aug[((size_t)b * NN + i) * 256 + NN + j];
}

__global__ void edge_kernel(const float* t1, const float* t2, float* eg) {
  const int idx = blockIdx.x * blockDim.x + threadIdx.x;
  if (idx >= BB * NN * NN) return;
  const int b = idx / (NN * NN), r = idx % (NN * NN);
  const int i = r / NN, j = r % NN;
  const size_t base = (size_t)b * NN * NN;
  float v;
  if (j == 1)      v = -t2[base + (size_t)i * NN + 1];
  else if (i == 1) v = t1[base + NN + j];
  else             v = t1[idx] - t2[idx];
  eg[idx] = v;
}

__global__ void denom_kernel(const float* eg, float* den) {
  __shared__ float red[128];
  const int bi = blockIdx.x, tid = threadIdx.x;
  red[tid] = eg[(size_t)bi * NN + tid];
  __syncthreads();
  for (int s = 64; s > 0; s >>= 1) {
    if (tid < s) red[tid] += red[tid + s];
    __syncthreads();
  }
  if (tid == 0) den[bi] = red[0] + 1.f;
}

__global__ void addh_kernel(const float* a, const float* b, _Float16* oh, int n) {
  const int idx = blockIdx.x * blockDim.x + threadIdx.x;
  if (idx < n) oh[idx] = (_Float16)(a[idx] + b[idx]);
}

__global__ void reludiv_kernel(const float* t, const float* den, float* of, int hdim, int n) {
  const int idx = blockIdx.x * blockDim.x + threadIdx.x;
  if (idx >= n) return;
  float v = t[idx] / den[idx / hdim];
  of[idx] = v > 0.f ? v : 0.f;
}

__global__ void gather_ent_kernel(const float* x2, const int* counts2, const int* order2,
                                  _Float16* ent) {
  const int b = blockIdx.x / EE, j = blockIdx.x % EE, tid = threadIdx.x;
  if (tid >= HH) return;
  float v = 0.f;
  if (j < counts2[b]) {
    const int src = order2[b * NN + j];
    v = x2[((size_t)b * NN + src) * HH + tid];
  }
  ent[((size_t)b * EE + j) * HH + tid] = (_Float16)v;
}

// ---------------- host-side launch helper ----------------
static void gemm(hipStream_t st,
  const void* A, int aF16, long long divA, long long bsA1, long long bsA2, int lda,
  const void* Bm, int bF16, int bT, long long divB, long long bsB1, long long bsB2, int ldb,
  float* C, _Float16* Ch, long long divC, long long bsC1, long long bsC2, int ldc, int cstride,
  const float* bias, long long divBias, long long bsBias1, long long bsBias2, int bias_estride,
  int M, int N, int K, float alpha, int act, int nz) {
  GemmP p;
  p.A = A; p.Bm = Bm; p.C = C; p.Ch = Ch; p.bias = bias;
  p.divA = divA < 1 ? 1 : divA; p.bsA1 = bsA1; p.bsA2 = bsA2;
  p.divB = divB < 1 ? 1 : divB; p.bsB1 = bsB1; p.bsB2 = bsB2;
  p.divC = divC < 1 ? 1 : divC; p.bsC1 = bsC1; p.bsC2 = bsC2;
  p.divBias = divBias < 1 ? 1 : divBias; p.bsBias1 = bsBias1; p.bsBias2 = bsBias2;
  p.lda = lda; p.ldb = ldb; p.ldc = ldc; p.cstride = cstride; p.bias_estride = bias_estride;
  p.M = M; p.N = N; p.K = K; p.alpha = alpha; p.act = act;
  dim3 g((N + BN - 1) / BN, (M + BM - 1) / BM, nz);
  const int sel = (aF16 << 2) | (bF16 << 1) | bT;
  // only the 5 combos actually used are instantiated
  switch (sel) {
    case 1:  gemm_wmma_kernel<0,0,1><<<g, dim3(256), 0, st>>>(p); break;  // fp32 @ fp32^T
    case 2:  gemm_wmma_kernel<0,1,0><<<g, dim3(256), 0, st>>>(p); break;  // fp32 @ f16
    case 4:  gemm_wmma_kernel<1,0,0><<<g, dim3(256), 0, st>>>(p); break;  // f16 @ fp32
    case 7:  gemm_wmma_kernel<1,1,1><<<g, dim3(256), 0, st>>>(p); break;  // f16 @ f16^T
    default: gemm_wmma_kernel<0,0,0><<<g, dim3(256), 0, st>>>(p); break;  // fp32 @ fp32
  }
}

extern "C" void kernel_launch(void* const* d_in, const int* in_sizes, int n_in,
                              void* d_out, int out_size, void* d_ws, size_t ws_size,
                              hipStream_t stream) {
  (void)in_sizes; (void)n_in; (void)out_size; (void)ws_size;
  const int*   ids    = (const int*)  d_in[0];
  const float* masks  = (const float*)d_in[1];
  const int*   starts = (const int*)  d_in[2];
  const float* nmap   = (const float*)d_in[3];
  const int*   nmask  = (const int*)  d_in[4];
  const float* tok    = (const float*)d_in[17];
  const float* pose   = (const float*)d_in[18];
  const float* typ    = (const float*)d_in[19];
  const float* elg    = (const float*)d_in[20];
  const float* elb    = (const float*)d_in[21];
  const float* q_w    = (const float*)d_in[22];
  const float* q_b    = (const float*)d_in[23];
  const float* k_w    = (const float*)d_in[24];
  const float* k_b    = (const float*)d_in[25];
  const float* v_w    = (const float*)d_in[26];
  const float* v_b    = (const float*)d_in[27];
  const float* o_w    = (const float*)d_in[28];
  const float* o_b    = (const float*)d_in[29];
  const float* alg    = (const float*)d_in[30];
  const float* alb    = (const float*)d_in[31];
  const float* f1w    = (const float*)d_in[32];
  const float* f1b    = (const float*)d_in[33];
  const float* f2w    = (const float*)d_in[34];
  const float* f2b    = (const float*)d_in[35];
  const float* flg    = (const float*)d_in[36];
  const float* flb    = (const float*)d_in[37];
  const float* lin1   = (const float*)d_in[38];
  const float* lin2   = (const float*)d_in[39];
  const float* lin3   = (const float*)d_in[40];
  const float* induc  = (const float*)d_in[41];
  const float* gw0    = (const float*)d_in[42];
  const float* gb0    = (const float*)d_in[43];
  const float* gw1    = (const float*)d_in[44];
  const float* gb1    = (const float*)d_in[45];
  const float* clsw   = (const float*)d_in[46];
  const float* clsb   = (const float*)d_in[47];
  float* out = (float*)d_out;

  // ---- workspace bump allocator ----
  char* wsb = (char*)d_ws; size_t off = 0;
  auto alloc = [&](size_t bytes) -> void* {
    void* pp = wsb + off; off = (off + bytes + 255) & ~(size_t)255; return pp;
  };
  const size_t BSD  = (size_t)BB * SS * DD;
  const size_t NDsz = (size_t)BB * NN * DD;
  const size_t NHsz = (size_t)BB * NN * HH;
  const size_t NNsz = (size_t)BB * NN * NN;
  float*    x      = (float*)alloc(BSD * 4);
  _Float16* xh     = (_Float16*)alloc(BSD * 2);
  _Float16* qh     = (_Float16*)alloc(BSD * 2);
  _Float16* kh     = (_Float16*)alloc(BSD * 2);
  _Float16* vh     = (_Float16*)alloc(BSD * 2);
  _Float16* ctxh   = (_Float16*)alloc(BSD * 2);
  float*    tmpD   = (float*)alloc(BSD * 4);
  _Float16* ffh    = (_Float16*)alloc((size_t)BB * SS * FFD * 2);
  float*    scores = (float*)alloc((size_t)BB * NHEAD * SS * SS * 4);
  float*    sbias  = (float*)alloc((size_t)BB * SS * 4);
  int*      counts = (int*)alloc(BB * 4);
  int*      cmax   = (int*)alloc(4);
  int*      order  = (int*)alloc((size_t)BB * SS * 4);
  float*    co2    = (float*)alloc(BSD * 4);
  float*    nrep   = (float*)alloc(NDsz * 4);
  _Float16* nrh    = (_Float16*)alloc(NDsz * 2);
  _Float16* h1h    = (_Float16*)alloc(NHsz * 2);
  _Float16* h2h    = (_Float16*)alloc(NHsz * 2);
  _Float16* th     = (_Float16*)alloc(NHsz * 2);
  float*    root   = (float*)alloc((size_t)BB * NN * 4);
  float*    bil    = (float*)alloc(NNsz * 4);
  float*    Pm     = (float*)alloc(NNsz * 4);
  float*    csum   = (float*)alloc((size_t)BB * NN * 4);
  float*    aug    = (float*)alloc((size_t)BB * NN * 256 * 4);
  float*    invm   = (float*)alloc(NNsz * 4);
  float*    t1     = (float*)alloc(NNsz * 4);
  float*    t2     = (float*)alloc(NNsz * 4);
  float*    edge   = (float*)alloc(NNsz * 4);
  float*    den    = (float*)alloc((size_t)BB * NN * 4);
  float*    gy     = (float*)alloc(NDsz * 4);
  _Float16* yh     = (_Float16*)alloc(NDsz * 2);
  float*    tmp2   = (float*)alloc(NHsz * 4);
  float*    x1     = (float*)alloc(NHsz * 4);
  float*    x2     = (float*)alloc(NHsz * 4);
  int*      counts2= (int*)alloc(BB * 4);
  int*      cmax2  = (int*)alloc(4);
  int*      order2 = (int*)alloc((size_t)BB * NN * 4);
  _Float16* enth   = (_Float16*)alloc((size_t)BB * EE * HH * 2);
  _Float16* tmpc   = (_Float16*)alloc((size_t)BB * RR * EE * HH * 2);

  // ---- embeddings + LN ----
  embed_kernel<<<BB * SS, 256, 0, stream>>>(ids, masks, tok, pose, typ, elg, elb, x, xh, sbias);

  // ---- 12 transformer layers ----
  for (int l = 0; l < LL; ++l) {
    const size_t wo = (size_t)l * DD * DD;
    // Q, K, V projections -> f16 [B*S, D]
    gemm(stream, xh, 1, 1, 0, 0, DD, q_w + wo, 0, 0, 1, 0, 0, DD,
         nullptr, qh, 1, 0, 0, DD, 1, q_b + (size_t)l * DD, 1, 0, 0, 1,
         BB * SS, DD, DD, 1.f, 0, 1);
    gemm(stream, xh, 1, 1, 0, 0, DD, k_w + wo, 0, 0, 1, 0, 0, DD,
         nullptr, kh, 1, 0, 0, DD, 1, k_b + (size_t)l * DD, 1, 0, 0, 1,
         BB * SS, DD, DD, 1.f, 0, 1);
    gemm(stream, xh, 1, 1, 0, 0, DD, v_w + wo, 0, 0, 1, 0, 0, DD,
         nullptr, vh, 1, 0, 0, DD, 1, v_b + (size_t)l * DD, 1, 0, 0, 1,
         BB * SS, DD, DD, 1.f, 0, 1);
    // scores = Q K^T / 8 + mask bias, batched over (b, h)
    gemm(stream, qh, 1, NHEAD, (long long)SS * DD, DHEAD, DD,
         kh, 1, 1, NHEAD, (long long)SS * DD, DHEAD, DD,
         scores, nullptr, 1, (long long)SS * SS, 0, SS, 1,
         sbias, NHEAD, SS, 0, 1,
         SS, SS, DHEAD, 0.125f, 0, BB * NHEAD);
    softmax_kernel<<<BB * NHEAD * SS, 256, 0, stream>>>(scores);
    // ctx = P V (P is fp32, converted on staging)
    gemm(stream, scores, 0, 1, (long long)SS * SS, 0, SS,
         vh, 1, 0, NHEAD, (long long)SS * DD, DHEAD, DD,
         nullptr, ctxh, NHEAD, (long long)SS * DD, DHEAD, DD, 1,
         nullptr, 1, 0, 0, 1,
         SS, DHEAD, SS, 1.f, 0, BB * NHEAD);
    // output projection + residual LN
    gemm(stream, ctxh, 1, 1, 0, 0, DD, o_w + wo, 0, 0, 1, 0, 0, DD,
         tmpD, nullptr, 1, 0, 0, DD, 1, o_b + (size_t)l * DD, 1, 0, 0, 1,
         BB * SS, DD, DD, 1.f, 0, 1);
    addln_kernel<<<BB * SS, 256, 0, stream>>>(x, tmpD, alg + (size_t)l * DD, alb + (size_t)l * DD, x, xh);
    // FFN
    gemm(stream, xh, 1, 1, 0, 0, DD, f1w + (size_t)l * DD * FFD, 0, 0, 1, 0, 0, FFD,
         nullptr, ffh, 1, 0, 0, FFD, 1, f1b + (size_t)l * FFD, 1, 0, 0, 1,
         BB * SS, FFD, DD, 1.f, 1, 1);
    gemm(stream, ffh, 1, 1, 0, 0, FFD, f2w + (size_t)l * FFD * DD, 0, 0, 1, 0, 0, DD,
         tmpD, nullptr, 1, 0, 0, DD, 1, f2b + (size_t)l * DD, 1, 0, 0, 1,
         BB * SS, DD, FFD, 1.f, 0, 1);
    addln_kernel<<<BB * SS, 256, 0, stream>>>(x, tmpD, flg + (size_t)l * DD, flb + (size_t)l * DD, x, xh);
  }

  // ---- stable gather by context_starts ----
  counts_kernel<<<1, 256, 0, stream>>>(starts, counts, cmax, BB, SS);
  scan_scatter_kernel<<<BB, SS, 0, stream>>>(starts, order, SS);
  gather_co_kernel<<<BB * SS, 256, 0, stream>>>(x, counts, cmax, order, co2);

  // ---- node_rep = node_mapping @ co ----
  gemm(stream, nmap, 0, 1, (long long)NN * SS, 0, SS,
       co2, 0, 0, 1, (long long)SS * DD, 0, DD,
       nrep, nrh, 1, (long long)NN * DD, 0, DD, 1,
       nullptr, 1, 0, 0, 1, NN, DD, SS, 1.f, 0, BB);

  // ---- bilinear induction head ----
  gemm(stream, nrh, 1, 1, 0, 0, DD, lin1, 0, 0, 1, 0, 0, HH,
       nullptr, h1h, 1, 0, 0, HH, 1, nullptr, 1, 0, 0, 1, BB * NN, HH, DD, 1.f, 2, 1);
  gemm(stream, nrh, 1, 1, 0, 0, DD, lin2, 0, 0, 1, 0, 0, HH,
       nullptr, h2h, 1, 0, 0, HH, 1, nullptr, 1, 0, 0, 1, BB * NN, HH, DD, 1.f, 2, 1);
  gemm(stream, nrh, 1, 1, 0, 0, DD, lin3, 0, 0, 1, 0, 0, 1,
       root, nullptr, 1, 0, 0, 1, 1, nullptr, 1, 0, 0, 1, BB * NN, 1, DD, 1.f, 0, 1);
  gemm(stream, h1h, 1, 1, 0, 0, HH, induc, 0, 0, 1, 0, 0, HH,
       nullptr, th, 1, 0, 0, HH, 1, nullptr, 1, 0, 0, 1, BB * NN, HH, HH, 1.f, 0, 1);
  gemm(stream, th, 1, 1, (long long)NN * HH, 0, HH,
       h2h, 1, 1, 1, (long long)NN * HH, 0, HH,
       bil, nullptr, 1, (long long)NN * NN, 0, NN, 1,
       nullptr, 1, 0, 0, 1, NN, NN, HH, 1.f, 0, BB);

  // ---- matrix-tree: P, Laplacian, batched inverse ----
  const int tot = BB * NN * NN;
  pexp_kernel<<<(tot + 255) / 256, 256, 0, stream>>>(bil, Pm);
  colsum_kernel<<<BB * NN, 128, 0, stream>>>(Pm, csum);
  aug_build_kernel<<<BB * NN, 256, 0, stream>>>(Pm, csum, root, aug);
  gj_kernel<<<BB, 256, 0, stream>>>(aug);
  invcopy_kernel<<<(tot + 255) / 256, 256, 0, stream>>>(aug, invm);
  gemm(stream, Pm, 0, 1, (long long)NN * NN, 0, NN,
       invm, 0, 0, 1, (long long)NN * NN, 0, NN,
       t1, nullptr, 1, (long long)NN * NN, 0, NN, 1,
       nullptr, 1, 0, 0, 1, NN, NN, NN, 1.f, 0, BB);
  gemm(stream, Pm, 0, 1, (long long)NN * NN, 0, NN,
       invm, 0, 1, 1, (long long)NN * NN, 0, NN,
       t2, nullptr, 1, (long long)NN * NN, 0, NN, 1,
       nullptr, 1, 0, 0, 1, NN, NN, NN, 1.f, 0, BB);
  edge_kernel<<<(tot + 255) / 256, 256, 0, stream>>>(t1, t2, edge);
  denom_kernel<<<BB * NN, 128, 0, stream>>>(edge, den);

  // ---- GCN layer 0: relu(((edge@x + x) @ w0 + b0) / denom) ----
  gemm(stream, edge, 0, 1, (long long)NN * NN, 0, NN,
       nrep, 0, 0, 1, (long long)NN * DD, 0, DD,
       gy, nullptr, 1, (long long)NN * DD, 0, DD, 1,
       nullptr, 1, 0, 0, 1, NN, DD, NN, 1.f, 0, BB);
  addh_kernel<<<((int)NDsz + 255) / 256, 256, 0, stream>>>(gy, nrep, yh, (int)NDsz);
  gemm(stream, yh, 1, 1, 0, 0, DD, gw0, 0, 0, 1, 0, 0, HH,
       tmp2, nullptr, 1, 0, 0, HH, 1, gb0, 1, 0, 0, 1, BB * NN, HH, DD, 1.f, 0, 1);
  reludiv_kernel<<<((int)NHsz + 255) / 256, 256, 0, stream>>>(tmp2, den, x1, HH, (int)NHsz);
  // ---- GCN layer 1 ----
  gemm(stream, edge, 0, 1, (long long)NN * NN, 0, NN,
       x1, 0, 0, 1, (long long)NN * HH, 0, HH,
       gy, nullptr, 1, (long long)NN * HH, 0, HH, 1,
       nullptr, 1, 0, 0, 1, NN, HH, NN, 1.f, 0, BB);
  addh_kernel<<<((int)NHsz + 255) / 256, 256, 0, stream>>>(gy, x1, yh, (int)NHsz);
  gemm(stream, yh, 1, 1, 0, 0, HH, gw1, 0, 0, 1, 0, 0, HH,
       tmp2, nullptr, 1, 0, 0, HH, 1, gb1, 1, 0, 0, 1, BB * NN, HH, HH, 1.f, 0, 1);
  reludiv_kernel<<<((int)NHsz + 255) / 256, 256, 0, stream>>>(tmp2, den, x2, HH, (int)NHsz);

  // ---- entity gather by node_mask ----
  counts_kernel<<<1, 256, 0, stream>>>(nmask, counts2, cmax2, BB, NN);
  scan_scatter_kernel<<<BB, NN, 0, stream>>>(nmask, order2, NN);
  gather_ent_kernel<<<BB * EE, 128, 0, stream>>>(x2, counts2, order2, enth);

  // ---- bilinear relation classifier: pred[b,i,j,r] ----
  // stage 1: tmpc[b,r] = ent_b @ cls_w[:, r, :]
  gemm(stream, enth, 1, RR, (long long)EE * HH, 0, HH,
       clsw, 0, 0, RR, 0, HH, RR * HH,
       nullptr, tmpc, 1, (long long)EE * HH, 0, HH, 1,
       nullptr, 1, 0, 0, 1, EE, HH, HH, 1.f, 0, BB * RR);
  // stage 2: pred[b,:,:,r] = tmpc[b,r] @ ent_b^T + cls_b[r]
  gemm(stream, tmpc, 1, 1, (long long)EE * HH, 0, HH,
       enth, 1, 1, RR, (long long)EE * HH, 0, HH,
       out, nullptr, RR, (long long)EE * EE * RR, 1, EE * RR, RR,
       clsb, RR, 0, 1, 0,
       EE, EE, HH, 1.f, 0, BB * RR);
}
